// BaseRNN_31473520345791
// MI455X (gfx1250) — matrix operand order repaired and checked by reading, not compiled, not run on hardware
//
#include <hip/hip_runtime.h>
#include <hip/hip_bf16.h>
#include <cstdint>

typedef __attribute__((ext_vector_type(16))) __bf16 v16bf;
typedef __attribute__((ext_vector_type(8)))  __bf16 v8bf;
typedef __attribute__((ext_vector_type(8)))  float  v8f;

// exact parameter type family for the async-to-LDS builtin (per hipcc diag):
typedef __attribute__((__vector_size__(4 * sizeof(int)))) int vs4i;
typedef vs4i __attribute__((address_space(1))) gvs4i;   // global
typedef vs4i __attribute__((address_space(3))) lvs4i;   // LDS

#define B_  128
#define T_  512
#define I_  256
#define H_  768
#define G4_ 3072   // 4*H

#define KCHUNK      256                       // K elements staged per chunk
#define STAGE_BYTES (KCHUNK * 64 * 2)         // 4 gates * 16 cols * bf16 = 32KB
#define GATE_BYTES  (KCHUNK * 16 * 2)         // 8192 B per gate within a stage

#if defined(__has_builtin)
#if __has_builtin(__builtin_amdgcn_global_load_async_to_lds_b128) && \
    __has_builtin(__builtin_amdgcn_s_wait_asynccnt)
#define USE_ASYNC_LDS 1
#endif
#endif
#ifndef USE_ASYNC_LDS
#define USE_ASYNC_LDS 0
#endif

// ---------------------------------------------------------------- conversion
__global__ __launch_bounds__(256) void cvt_f32_bf16(const float* __restrict__ in,
                                                    __bf16* __restrict__ out, long n) {
  long i = (long)blockIdx.x * blockDim.x + threadIdx.x;
  long stride = (long)gridDim.x * blockDim.x;
  for (; i < n; i += stride) out[i] = (__bf16)in[i];
}

// ---------------------------------------------------------------- helpers
__device__ __forceinline__ v8f wmma_bf16(v16bf a, v16bf b, v8f c) {
  return __builtin_amdgcn_wmma_f32_16x16x32_bf16(false, a, false, b, (short)0, c,
                                                 false, false);
}
__device__ __forceinline__ float fast_sigmoid(float x) {
  return 1.0f / (1.0f + __expf(-x));
}
__device__ __forceinline__ float fast_tanh(float x) {
  float e = __expf(2.0f * x);
  return (e - 1.0f) / (e + 1.0f);
}

// Stage one K=256 chunk of B ([4 gates][256 K][16 cols] bf16 = 32KB) into LDS.
// bsrc points at (row k0, col n0) of the bf16 weight matrix (row stride G4_).
// LDS layout: loff = g*GATE_BYTES + kk*32 + half*16.
__device__ __forceinline__ void stage_chunk(const __bf16* __restrict__ bsrc,
                                            char* __restrict__ sdst, int tid) {
#pragma unroll
  for (int i = 0; i < 8; ++i) {
    const int q    = tid + i * 256;   // 2048 16B transfers
    const int half = q & 1;
    const int row  = q >> 1;          // 0..1023
    const int g    = row >> 8;        // gate 0..3
    const int kk   = row & (KCHUNK - 1);
    const __bf16* gp = bsrc + (size_t)kk * G4_ + g * H_ + half * 8;
    char* lp = sdst + g * GATE_BYTES + kk * 32 + half * 16;
#if USE_ASYNC_LDS
    __builtin_amdgcn_global_load_async_to_lds_b128(
        (gvs4i*)(uintptr_t)gp,
        (lvs4i*)(uint32_t)(uintptr_t)lp,
        0, 0);
#else
    *(v8bf*)lp = *(const v8bf*)gp;
#endif
  }
}

__device__ __forceinline__ void stage_wait() {
#if USE_ASYNC_LDS
  __builtin_amdgcn_s_wait_asynccnt(0);
#endif
  __syncthreads();
}

// ---------------------------------------------------------------- LSTM step
// Block = one 16-wide hidden-column tile (n0 = blockIdx.x*16); its 8 waves
// cover the 8 batch tiles, so all waves consume the SAME B slice -> stage B
// through LDS once per block (8x less L2 traffic). Each wave holds FOUR f32
// accumulator tiles (i,f,g,o) split into TWO sets (even/odd K-groups) to
// halve the dependent-WMMA chain. K = I (x@W) + H (h@U) = 1024 = 4 chunks.
__global__ __launch_bounds__(256) void lstm_step(
    const __bf16* __restrict__ xbf,     // [B,T,I]  bf16
    const __bf16* __restrict__ Wbf,     // [I,4H]   bf16
    const __bf16* __restrict__ Ubf,     // [H,4H]   bf16
    const __bf16* __restrict__ hbf_in,  // [B,H]    bf16 (h_{t-1})
    __bf16* __restrict__ hbf_out,       // [B,H]    bf16 (h_t, for next step)
    const float* __restrict__ bias,     // [4H]
    const float* __restrict__ cprev,    // base ptr for c_{t-1}
    long cstride,                       // row stride of cprev (H or T*H)
    float* __restrict__ hs_t,           // d_out       + t*H, row stride T*H
    float* __restrict__ cs_t,           // d_out + BTH + t*H, row stride T*H
    int t) {
  __shared__ __align__(32) char smem[2 * STAGE_BYTES];   // 64KB double buffer

  const int tid   = threadIdx.x;
  const int lane  = tid & 31;
  const int n0    = blockIdx.x * 16;          // hidden-column tile
  const int m0    = (tid >> 5) * 16;          // batch tile = wave id
  const int laneq = lane & 15;
  const int kofA  = (lane >> 4) * 8;          // A fragment K sub-offset
  const int mbase = m0 + (lane >> 4) * 8;
  const int n     = n0 + laneq;

  // ---- hoisted epilogue operands (latency hidden under the GEMM)
  const float bI = bias[n];
  const float bF = bias[n + H_];
  const float bG = bias[n + 2 * H_];
  const float bO = bias[n + 3 * H_];
  float cp[8];
#pragma unroll
  for (int j = 0; j < 8; ++j)
    cp[j] = cprev[(size_t)(mbase + j) * (size_t)cstride + n];

  v8f accI0 = {}, accF0 = {}, accG0 = {}, accO0 = {};
  v8f accI1 = {}, accF1 = {}, accG1 = {}, accO1 = {};

  // prologue: stage chunk 0 (the W/x chunk)
  stage_chunk(Wbf + n0, smem, tid);
  stage_wait();

  for (int c = 0; c < 4; ++c) {
    char* cur = smem + (c & 1) * STAGE_BYTES;
    // stage chunk c+1 (always a U chunk) into the other buffer
    if (c < 3)
      stage_chunk(Ubf + (size_t)c * KCHUNK * G4_ + n0,
                  smem + ((c + 1) & 1) * STAGE_BYTES, tid);

    const __bf16* aBase =
        (c == 0) ? (xbf + ((size_t)(m0 + laneq) * T_ + t) * I_ + kofA)
                 : (hbf_in + (size_t)(m0 + laneq) * H_ + (c - 1) * KCHUNK + kofA);

    // 8 K-groups per chunk; alternate accumulator sets for ILP
#pragma unroll
    for (int j = 0; j < 8; ++j) {
      const int kk0 = j * 32;
      v8bf alo = *(const v8bf*)(aBase + kk0);
      v8bf ahi = *(const v8bf*)(aBase + kk0 + 16);
      v16bf a = __builtin_shufflevector(alo, ahi, 0, 1, 2, 3, 4, 5, 6, 7,
                                        8, 9, 10, 11, 12, 13, 14, 15);
      const char* bl = cur + (kk0 + kofA) * 32 + laneq * 2;   // col n = laneq, K in table order
      v16bf b0, b1, b2, b3;
#pragma unroll
      for (int s = 0; s < 16; ++s) {
        const int ko = ((s < 8) ? s : s + 8) * 32;
        b0[s] = *(const __bf16*)(bl + ko);
        b1[s] = *(const __bf16*)(bl + ko + GATE_BYTES);
        b2[s] = *(const __bf16*)(bl + ko + 2 * GATE_BYTES);
        b3[s] = *(const __bf16*)(bl + ko + 3 * GATE_BYTES);
      }
      if (j & 1) {
        accI1 = wmma_bf16(a, b0, accI1);
        accF1 = wmma_bf16(a, b1, accF1);
        accG1 = wmma_bf16(a, b2, accG1);
        accO1 = wmma_bf16(a, b3, accO1);
      } else {
        accI0 = wmma_bf16(a, b0, accI0);
        accF0 = wmma_bf16(a, b1, accF0);
        accG0 = wmma_bf16(a, b2, accG0);
        accO0 = wmma_bf16(a, b3, accO0);
      }
    }
    if (c < 3) stage_wait();   // next buffer ready; all readers past cur
  }

  // combine accumulator sets
  v8f accI = accI0 + accI1;
  v8f accF = accF0 + accF1;
  v8f accG = accG0 + accG1;
  v8f accO = accO0 + accO1;

  // ---- epilogue: gates -> cell/hidden update
  const size_t ostride = (size_t)T_ * H_;
#pragma unroll
  for (int j = 0; j < 8; ++j) {
    const int m = mbase + j;                       // batch index
    float ig = fast_sigmoid(accI[j] + bI);
    float fg = fast_sigmoid(accF[j] + bF);
    float gg = fast_tanh(accG[j] + bG);
    float og = fast_sigmoid(accO[j] + bO);
    float cn = fg * cp[j] + ig * gg;
    float hn = og * fast_tanh(cn);
    hs_t[(size_t)m * ostride + n] = hn;
    cs_t[(size_t)m * ostride + n] = cn;
    hbf_out[(size_t)m * H_ + n] = (__bf16)hn;      // feed next step's A matrix
  }
}

// ---------------------------------------------------------------- launcher
extern "C" void kernel_launch(void* const* d_in, const int* in_sizes, int n_in,
                              void* d_out, int out_size, void* d_ws, size_t ws_size,
                              hipStream_t stream) {
  const float* x  = (const float*)d_in[0];  // [B,T,I]
  const float* h0 = (const float*)d_in[1];  // [B,H]
  const float* c0 = (const float*)d_in[2];  // [B,H]
  const float* W  = (const float*)d_in[3];  // [I,4H]
  const float* U  = (const float*)d_in[4];  // [H,4H]
  const float* b  = (const float*)d_in[5];  // [4H]

  float* hs = (float*)d_out;                          // [B,T,H]
  float* cs = hs + (size_t)B_ * T_ * H_;              // [B,T,H]

  // workspace layout (all offsets 256B aligned)
  char* ws = (char*)d_ws;
  __bf16* xbf  = (__bf16*)(ws);                       // 33,554,432 B
  __bf16* Wbf  = (__bf16*)(ws + 33554432);            //  1,572,864 B
  __bf16* Ubf  = (__bf16*)(ws + 35127296);            //  4,718,592 B
  __bf16* hbf0 = (__bf16*)(ws + 39845888);            //    196,608 B
  __bf16* hbf1 = (__bf16*)(ws + 40042496);            //    196,608 B

  // one-time fp32 -> bf16 conversion of x, W, U, h0
  cvt_f32_bf16<<<4096, 256, 0, stream>>>(x,  xbf, (long)B_ * T_ * I_);
  cvt_f32_bf16<<<768,  256, 0, stream>>>(W,  Wbf, (long)I_ * G4_);
  cvt_f32_bf16<<<2304, 256, 0, stream>>>(U,  Ubf, (long)H_ * G4_);
  cvt_f32_bf16<<<96,   256, 0, stream>>>(h0, hbf0, (long)B_ * H_);

  // one block per hidden-column tile; 8 waves = 8 batch tiles
  const int blocks = H_ / 16;   // 48
  for (int t = 0; t < T_; ++t) {
    const __bf16* hin = (t & 1) ? hbf1 : hbf0;
    __bf16* hout      = (t & 1) ? hbf0 : hbf1;
    const float* cprev = (t == 0) ? c0 : (cs + (size_t)(t - 1) * H_);
    long cstride       = (t == 0) ? (long)H_ : (long)T_ * H_;
    lstm_step<<<blocks, 256, 0, stream>>>(xbf, Wbf, Ubf, hin, hout, b,
                                          cprev, cstride,
                                          hs + (size_t)t * H_,
                                          cs + (size_t)t * H_, t);
  }
}
